// MultiheadAttention_67233418052114
// MI455X (gfx1250) — matrix operand order repaired and checked by reading, not compiled, not run on hardware
//
#include <hip/hip_runtime.h>

// ---------------------------------------------------------------------------
// MultiheadAttention on gfx1250 (MI455X), bf16 WMMA + TDM pipeline, round 4.
//   B=4, S=8192, E=1024, H=16, D=64 -> effective S'=512/head, out [4,512,1024] f32.
//
// Pipeline (~170 MB HBM total, bandwidth-bound):
//   1. weights -> bf16 (Wq|Wk|Wv stacked [192,1024]); biases concatenated.
//   2. fused QKV GEMM [32768,1024]x[1024,192]: A read as f32, converted to
//      bf16 in-register (x touches HBM exactly once).
//   3. V -> head-major K-contiguous [B,H,64,512] via LDS-tiled transpose.
//   4. flash attention per (b,h): K/V tiles DMA'd to LDS by the Tensor Data
//      Mover (tensor_load_to_lds, double buffered, s_wait_tensorcnt), scores
//      via WMMA, online softmax (row-max via wave32 shuffles, row-sum via a
//      P@ones WMMA), P transposed D->A layout through a per-wave LDS slab.
//   5. output projection [2048,1024]x[1024,1024]+bias -> f32 d_out.
// ---------------------------------------------------------------------------

typedef unsigned short u16;
typedef __attribute__((ext_vector_type(16))) __bf16 v16bf;
typedef __attribute__((ext_vector_type(8)))  float  v8f;

union Frag {
    v16bf v;
    uint4 q[2];
    u16   e[16];
};

__device__ __forceinline__ u16 f2bf(float f) {
    unsigned u = __float_as_uint(f);
    u += 0x7fffu + ((u >> 16) & 1u);   // round-to-nearest-even
    return (u16)(u >> 16);
}

// ---------------------------------------------------------------------------
// Tensor Data Mover: 2-D tile load Global->LDS (D# per ISA cdna5 §8.3/8.4).
// dims/strides in elements of data_size=2 bytes.
// This toolchain's builtin is the 6-arg form:
//   (uint32x4 g0, int32x8 g1, int32x4 g2, int32x4 g3, int32x8 pad, i32 cpol)
// ---------------------------------------------------------------------------
#if __has_builtin(__builtin_amdgcn_tensor_load_to_lds) && __has_builtin(__builtin_amdgcn_s_wait_tensorcnt)
#define HAS_TDM 1
typedef __attribute__((ext_vector_type(4))) unsigned int v4u;
typedef __attribute__((ext_vector_type(8))) int v8i;
typedef __attribute__((ext_vector_type(4))) int v4i;

__device__ __forceinline__ unsigned lds_off(const void* p) {
    return (unsigned)(unsigned long long)(__attribute__((address_space(3))) const char*)p;
}

__device__ __forceinline__ void tdm_load_2d(unsigned lds, unsigned long long gaddr,
                                            unsigned dim0, unsigned dim1,
                                            unsigned long long stride0,
                                            unsigned tile0, unsigned tile1)
{
    v4u g0;
    g0[0] = 1u;                                                    // count=1 (valid, user mode)
    g0[1] = lds;                                                   // lds_addr (bytes)
    g0[2] = (unsigned)(gaddr & 0xffffffffull);                     // global_addr[31:0]
    g0[3] = (unsigned)((gaddr >> 32) & 0x01ffffffull) | (2u << 30);// addr[56:32] | type=2
    v8i g1;
    g1[0] = (int)(1u << 16);                                       // data_size=1 (2B), mask=0
    g1[1] = (int)((dim0 & 0xffffu) << 16);                         // tensor_dim0[15:0]
    g1[2] = (int)(((dim0 >> 16) & 0xffffu) | ((dim1 & 0xffffu) << 16));
    g1[3] = (int)(((dim1 >> 16) & 0xffffu) | (tile0 << 16));       // tile_dim0
    g1[4] = (int)(tile1 & 0xffffu);                                // tile_dim1 | tile_dim2=0
    g1[5] = (int)(stride0 & 0xffffffffull);                        // dim0_stride[31:0]
    g1[6] = (int)((stride0 >> 32) & 0xffffull);                    // dim0_stride[47:32] | dim1_stride lo=0
    g1[7] = 0;                                                     // dim1_stride hi = 0
    v4i z4 = {0, 0, 0, 0};
    v8i z8 = {0, 0, 0, 0, 0, 0, 0, 0};
    __builtin_amdgcn_tensor_load_to_lds(g0, g1, z4, z4, z8, 0);
}
#else
#define HAS_TDM 0
#endif

// ---------------------------------------------------------------------------
// f32 -> bf16 conversion (weights only)
// ---------------------------------------------------------------------------
__global__ void f32_to_bf16(const float* __restrict__ in, u16* __restrict__ out, long n) {
    long i = (long)blockIdx.x * blockDim.x + threadIdx.x;
    long stride = (long)gridDim.x * blockDim.x;
    for (; i < n; i += stride) out[i] = f2bf(in[i]);
}

__global__ void concat_bias(const float* __restrict__ bq, const float* __restrict__ bk,
                            const float* __restrict__ bv, float* __restrict__ o) {
    int i = threadIdx.x;   // 192 threads
    o[i] = (i < 64) ? bq[i] : (i < 128 ? bk[i - 64] : bv[i - 128]);
}

// ---------------------------------------------------------------------------
// V gather/transpose via LDS tile (coalesced on both sides):
//   vg[((b*16+h)*64 + d)*512 + s] = qkv[(b*8192 + s*16 + h)*192 + 128 + d]
// Grid: 256 blocks = 64 heads x 4 s-tiles of 128.
// ---------------------------------------------------------------------------
__global__ void __launch_bounds__(256)
gather_v(const u16* __restrict__ qkv, u16* __restrict__ vg) {
    __shared__ u16 tile[64][128];
    const int z  = blockIdx.x >> 2;          // b*16 + h
    const int s0 = (blockIdx.x & 3) * 128;
    const int b = z >> 4, h = z & 15;
    const u16* src = qkv + ((long)b * 8192 + h) * 192 + 128;
    const int d  = threadIdx.x & 63;
    const int sr = threadIdx.x >> 6;         // 0..3
    #pragma unroll 4
    for (int it = 0; it < 32; ++it) {
        int s = s0 + it * 4 + sr;
        tile[d][s - s0] = src[(long)s * 3072 + d];     // 64 consecutive d: coalesced
    }
    __syncthreads();
    const int sl = threadIdx.x & 127;
    const int dr = threadIdx.x >> 7;         // 0..1
    u16* dst = vg + (long)z * 64 * 512 + s0;
    #pragma unroll 4
    for (int it = 0; it < 32; ++it) {
        int dd = dr + it * 2;
        dst[(long)dd * 512 + sl] = tile[dd][sl];       // 128 consecutive s: coalesced
    }
}

// ---------------------------------------------------------------------------
// Generic bf16 WMMA GEMM:  Out = scale * (A @ B) + bias
//   A : A(m,k) = A[m*lda + k]; bf16, or f32 converted in-register (A_F32)
//   B : bf16, B(k,n) = Bm[n*ldb + k]              (K-contiguous per column)
//   Out: f32 or bf16, Out(m,n) = Out[m*ldo + n]
// 256 threads = 8 waves; block tile 128x64; wave tile 32x32 (4 WMMA/K-step).
// ---------------------------------------------------------------------------
template <bool OUT_BF, bool A_F32>
__global__ void __launch_bounds__(256)
gemm_bf16_wmma(const void* __restrict__ Avoid, long lda,
               const u16* __restrict__ Bm, long ldb,
               const float* __restrict__ bias, float scale,
               void* __restrict__ Out, long ldo,
               int K)
{
    const int lane = threadIdx.x & 31;
    const int wave = threadIdx.x >> 5;
    const int wm = blockIdx.x * 128 + (wave >> 1) * 32;
    const int wn = blockIdx.y * 64  + (wave & 1)  * 32;
    const int r  = lane & 15;
    const int hi = lane >> 4;

    v8f acc[2][2] = {};

    for (int k0 = 0; k0 < K; k0 += 32) {
        Frag a[2], b[2];
        #pragma unroll
        for (int mt = 0; mt < 2; ++mt) {
            if (A_F32) {
                const float* p = (const float*)Avoid + (long)(wm + mt * 16 + r) * lda + k0 + hi * 8;
                float4 c0 = *(const float4*)(p);
                float4 c1 = *(const float4*)(p + 4);
                float4 c2 = *(const float4*)(p + 16);
                float4 c3 = *(const float4*)(p + 20);
                a[mt].e[0]  = f2bf(c0.x); a[mt].e[1]  = f2bf(c0.y);
                a[mt].e[2]  = f2bf(c0.z); a[mt].e[3]  = f2bf(c0.w);
                a[mt].e[4]  = f2bf(c1.x); a[mt].e[5]  = f2bf(c1.y);
                a[mt].e[6]  = f2bf(c1.z); a[mt].e[7]  = f2bf(c1.w);
                a[mt].e[8]  = f2bf(c2.x); a[mt].e[9]  = f2bf(c2.y);
                a[mt].e[10] = f2bf(c2.z); a[mt].e[11] = f2bf(c2.w);
                a[mt].e[12] = f2bf(c3.x); a[mt].e[13] = f2bf(c3.y);
                a[mt].e[14] = f2bf(c3.z); a[mt].e[15] = f2bf(c3.w);
            } else {
                const u16* p = (const u16*)Avoid + (long)(wm + mt * 16 + r) * lda + k0 + hi * 8;
                a[mt].q[0] = *(const uint4*)(p);
                a[mt].q[1] = *(const uint4*)(p + 16);
            }
        }
        #pragma unroll
        for (int nt = 0; nt < 2; ++nt) {
            const u16* p = Bm + (long)(wn + nt * 16 + r) * ldb + k0 + hi * 8;
            b[nt].q[0] = *(const uint4*)(p);
            b[nt].q[1] = *(const uint4*)(p + 16);
        }
        #pragma unroll
        for (int mt = 0; mt < 2; ++mt)
            #pragma unroll
            for (int nt = 0; nt < 2; ++nt)
                acc[mt][nt] = __builtin_amdgcn_wmma_f32_16x16x32_bf16(
                    false, a[mt].v, false, b[nt].v,
                    (short)0, acc[mt][nt], false, false);
    }

    #pragma unroll
    for (int mt = 0; mt < 2; ++mt) {
        #pragma unroll
        for (int nt = 0; nt < 2; ++nt) {
            const int col = wn + nt * 16 + r;
            const float bv = bias ? bias[col] : 0.0f;
            #pragma unroll
            for (int rr = 0; rr < 8; ++rr) {
                const int row = wm + mt * 16 + rr + hi * 8;
                const float val = acc[mt][nt][rr] * scale + bv;
                const long idx = (long)row * ldo + col;
                if (OUT_BF) ((u16*)Out)[idx]   = f2bf(val);
                else        ((float*)Out)[idx] = val;
            }
        }
    }
}

// ---------------------------------------------------------------------------
// Fused flash attention for one (b,h): softmax(Q K^T / 8) @ V.
//   qkv : [32768, 192] bf16 (Q cols 0..63, K cols 64..127)
//   vg  : [64 heads][64 d][512 s] bf16
//   ctx : [2048, 1024] bf16; head h -> columns h*64..h*64+63
// Grid (4,1,64); 256 threads = 8 waves; each wave owns 16 query rows.
// K/V tiles staged in LDS by TDM (wave 0 issues, double buffered); row sums
// computed with a P@ones WMMA; P transposed through a per-wave LDS slab.
// ---------------------------------------------------------------------------
__global__ void __launch_bounds__(256)
flash_attn(const u16* __restrict__ qkv, const u16* __restrict__ vg,
           u16* __restrict__ ctx)
{
    __shared__ u16 pbuf[8][16][32];          // per-wave P-tile slab (8 KB)
#if HAS_TDM
    __shared__ u16 kbuf[2][32][64];          // double-buffered K tile (8 KB)
    __shared__ u16 vbuf[2][64][32];          // double-buffered V tile (8 KB)
#endif

    const int z = blockIdx.z;                // b*16 + h
    const int b = z >> 4;
    const int h = z & 15;
    const int lane = threadIdx.x & 31;
    const int wave = threadIdx.x >> 5;
    const int r  = lane & 15;
    const int hi = lane >> 4;
    const int qbase = blockIdx.x * 128 + wave * 16;

    const u16* qp = qkv + ((long)b * 8192 + h) * 192;   // + row*3072 + d
    const u16* kp = qp + 64;                            // K column block
    const u16* vp = vg + (long)z * 64 * 512;            // + d*512 + s

    // Q fragments (16 x 64) kept in registers for the whole pass
    Frag qf[2];
    #pragma unroll
    for (int kf = 0; kf < 2; ++kf) {
        const u16* p = qp + (long)(qbase + r) * 3072 + kf * 32 + hi * 8;
        qf[kf].q[0] = *(const uint4*)(p);
        qf[kf].q[1] = *(const uint4*)(p + 16);
    }

    // all-ones B fragment: rowsum(P) = P @ ones via the matrix engine
    Frag onef;
    #pragma unroll
    for (int j = 0; j < 16; ++j) onef.e[j] = 0x3f80;    // bf16 1.0

    v8f acc[4] = {};                         // 16q x 64d output accumulator
    float m_run[8], l_run[8];
    #pragma unroll
    for (int i = 0; i < 8; ++i) { m_run[i] = -3.0e38f; l_run[i] = 0.0f; }

#if HAS_TDM
#define STAGE_KV(bi_, jb_)                                                    \
    do {                                                                      \
        tdm_load_2d(lds_off(&kbuf[(bi_)][0][0]),                              \
                    (unsigned long long)(kp + (long)(jb_) * 3072),            \
                    64u, 32u, 3072ull, 64u, 32u);                             \
        tdm_load_2d(lds_off(&vbuf[(bi_)][0][0]),                              \
                    (unsigned long long)(vp + (jb_)),                         \
                    32u, 64u, 512ull, 32u, 64u);                              \
    } while (0)
    if (wave == 0) STAGE_KV(0, 0);
#endif

    for (int i = 0; i < 16; ++i) {
        const int jb = i * 32;
        const u16* kblk; const u16* vblk; int ks, vs;
#if HAS_TDM
        const int bi = i & 1;
        if (wave == 0) {
            if (i + 1 < 16) {
                STAGE_KV((i + 1) & 1, (i + 1) * 32);
                __builtin_amdgcn_s_wait_tensorcnt(2);   // stage i complete
            } else {
                __builtin_amdgcn_s_wait_tensorcnt(0);
            }
        }
        __syncthreads();                                // tile visible to all waves
        kblk = &kbuf[bi][0][0]; ks = 64;
        vblk = &vbuf[bi][0][0]; vs = 32;
#else
        kblk = kp + (long)jb * 3072; ks = 3072;
        vblk = vp + jb;              vs = 512;
#endif

        // ---- scores S = (Q K^T) / 8 : 16 x 32 tile ----
        v8f s[2] = {};
        #pragma unroll
        for (int kf = 0; kf < 2; ++kf) {
            #pragma unroll
            for (int nt = 0; nt < 2; ++nt) {
                Frag bfr;
                const u16* p = kblk + (long)(nt * 16 + r) * ks + kf * 32 + hi * 8;
                bfr.q[0] = *(const uint4*)(p);
                bfr.q[1] = *(const uint4*)(p + 16);
                s[nt] = __builtin_amdgcn_wmma_f32_16x16x32_bf16(
                    false, qf[kf].v, false, bfr.v, (short)0, s[nt], false, false);
            }
        }
        #pragma unroll
        for (int nt = 0; nt < 2; ++nt)
            #pragma unroll
            for (int rr = 0; rr < 8; ++rr) s[nt][rr] *= 0.125f;

        // ---- online softmax: row max via 16-lane-half shuffles ----
        float mnew[8], alpha[8];
        #pragma unroll
        for (int rr = 0; rr < 8; ++rr) {
            float rm = fmaxf(s[0][rr], s[1][rr]);
            #pragma unroll
            for (int o = 1; o < 16; o <<= 1) rm = fmaxf(rm, __shfl_xor(rm, o, 32));
            mnew[rr]  = fmaxf(m_run[rr], rm);
            alpha[rr] = __expf(m_run[rr] - mnew[rr]);
            m_run[rr] = mnew[rr];
        }
        #pragma unroll
        for (int nt = 0; nt < 2; ++nt)
            #pragma unroll
            for (int rr = 0; rr < 8; ++rr) s[nt][rr] = __expf(s[nt][rr] - mnew[rr]);
        #pragma unroll
        for (int dt = 0; dt < 4; ++dt)
            #pragma unroll
            for (int rr = 0; rr < 8; ++rr) acc[dt][rr] *= alpha[rr];

        // ---- P tile: D layout -> row-major bf16 in this wave's LDS slab ----
        #pragma unroll
        for (int nt = 0; nt < 2; ++nt)
            #pragma unroll
            for (int rr = 0; rr < 8; ++rr)
                pbuf[wave][rr + 8 * hi][nt * 16 + r] = f2bf(s[nt][rr]);

        // ---- P as A fragment (16 x 32) from LDS (same-wave, in-order) ----
        Frag pf;
        {
            const u16* p = &pbuf[wave][r][hi * 8];
            pf.q[0] = *(const uint4*)(p);
            pf.q[1] = *(const uint4*)(p + 16);
        }

        // ---- row sums via matrix engine: lsum = P @ ones (replicated/half) ----
        v8f lsum = __builtin_amdgcn_wmma_f32_16x16x32_bf16(
            false, pf.v, false, onef.v, (short)0, v8f{}, false, false);
        #pragma unroll
        for (int rr = 0; rr < 8; ++rr) l_run[rr] = l_run[rr] * alpha[rr] + lsum[rr];

        // ---- acc += P @ V ----
        #pragma unroll
        for (int dt = 0; dt < 4; ++dt) {
            Frag vfr;
            const u16* p = vblk + (long)(dt * 16 + r) * vs + hi * 8;
            vfr.q[0] = *(const uint4*)(p);
            vfr.q[1] = *(const uint4*)(p + 16);
            acc[dt] = __builtin_amdgcn_wmma_f32_16x16x32_bf16(
                false, pf.v, false, vfr.v, (short)0, acc[dt], false, false);
        }
#if HAS_TDM
        __syncthreads();                                // done reading buf[bi]
#endif
    }

    // ---- epilogue: ctx[b*512 + q, h*64 + d] = acc / l ----
    u16* op = ctx + (long)(b * 512 + qbase) * 1024 + h * 64;
    #pragma unroll
    for (int rr = 0; rr < 8; ++rr) {
        const float inv = 1.0f / l_run[rr];
        #pragma unroll
        for (int dt = 0; dt < 4; ++dt)
            op[(long)(rr + 8 * hi) * 1024 + dt * 16 + r] = f2bf(acc[dt][rr] * inv);
    }
}

// ---------------------------------------------------------------------------
// Host-side orchestration
// ---------------------------------------------------------------------------
extern "C" void kernel_launch(void* const* d_in, const int* in_sizes, int n_in,
                              void* d_out, int out_size, void* d_ws, size_t ws_size,
                              hipStream_t stream)
{
    (void)in_sizes; (void)n_in; (void)out_size; (void)ws_size;

    const float* x  = (const float*)d_in[0];
    const float* Wq = (const float*)d_in[1];
    const float* bq = (const float*)d_in[2];
    const float* Wk = (const float*)d_in[3];
    const float* bk = (const float*)d_in[4];
    const float* Wv = (const float*)d_in[5];
    const float* bv = (const float*)d_in[6];
    const float* Wo = (const float*)d_in[7];
    const float* bo = (const float*)d_in[8];
    float* out = (float*)d_out;

    // Workspace layout (u16 units), ~24 MB total
    u16* wqkv_bf = (u16*)d_ws;                 // [192,1024]      = 196,608
    u16* wo_bf   = wqkv_bf + 196608;           // [1024,1024]     = 1,048,576
    u16* qkv_bf  = wo_bf   + 1048576;          // [32768,192]     = 6,291,456
    u16* vg_bf   = qkv_bf  + 6291456;          // [64,64,512]     = 2,097,152
    u16* ctx_bf  = vg_bf   + 2097152;          // [2048,1024]     = 2,097,152
    float* bqkv  = (float*)(ctx_bf + 2097152); // 192 f32

    // 1) weight staging
    f32_to_bf16<<<256,  256, 0, stream>>>(Wq, wqkv_bf,          65536);
    f32_to_bf16<<<256,  256, 0, stream>>>(Wk, wqkv_bf + 65536,  65536);
    f32_to_bf16<<<256,  256, 0, stream>>>(Wv, wqkv_bf + 131072, 65536);
    f32_to_bf16<<<1024, 256, 0, stream>>>(Wo, wo_bf,            1048576);
    concat_bias<<<1, 192, 0, stream>>>(bq, bk, bv, bqkv);

    // 2) fused QKV projection (A = x as f32, converted in-register)
    gemm_bf16_wmma<true, true><<<dim3(256, 3, 1), 256, 0, stream>>>(
        x, 1024,  wqkv_bf, 1024,  bqkv, 1.0f,  qkv_bf, 192,  1024);

    // 3) V -> head-major K-contiguous [B,H,64,512]
    gather_v<<<256, 256, 0, stream>>>(qkv_bf, vg_bf);

    // 4) fused attention -> ctx_bf [2048,1024]
    flash_attn<<<dim3(4, 1, 64), 256, 0, stream>>>(qkv_bf, vg_bf, ctx_bf);

    // 5) out = ctx @ Wo^T + bo -> f32 d_out
    gemm_bf16_wmma<false, false><<<dim3(16, 16, 1), 256, 0, stream>>>(
        ctx_bf, 1024,  wo_bf, 1024,  bo, 1.0f,  out, 1024,  1024);
}
